// vanilla_TLA_46273977647486
// MI455X (gfx1250) — matrix operand order repaired and checked by reading, not compiled, not run on hardware
//
#include <hip/hip_runtime.h>
#include <hip/hip_bf16.h>
#include <math.h>

// ---- problem constants (from the reference) ----
#define BATCH   256
#define SEQ_L   12
#define VOCAB   10000
#define HID     1024          // K of the big GEMM
#define NOUT    120000        // LO*VO = N of the big GEMM
#define NTILES  (NOUT / 16)   // 7500
#define MTILES  (BATCH / 16)  // 16

typedef _Float16 v16h __attribute__((ext_vector_type(16)));
typedef _Float16 v8h  __attribute__((ext_vector_type(8)));
typedef float    v8f  __attribute__((ext_vector_type(8)));
typedef float    v4f  __attribute__((ext_vector_type(4)));

// ------------------------------------------------------------------
// Kernel 1: hid[b,h] = tanh( sum_l W1[h, l*V + X[b,l]] + b1[h] ) -> fp16
// One block per batch row, 1024 threads = one thread per h.
// ------------------------------------------------------------------
__global__ void __launch_bounds__(HID)
nnlm_hidden_kernel(const int* __restrict__ X,
                   const float* __restrict__ W1,
                   const float* __restrict__ b1,
                   _Float16* __restrict__ hidH)
{
    const int b = blockIdx.x;
    const int h = threadIdx.x;

    __shared__ int xs[SEQ_L];
    if (threadIdx.x < SEQ_L) xs[threadIdx.x] = X[b * SEQ_L + threadIdx.x];
    __syncthreads();

    const float* w1row = W1 + (size_t)h * (size_t)(VOCAB * SEQ_L);
    float s = b1[h];
#pragma unroll
    for (int l = 0; l < SEQ_L; ++l) {
        s += w1row[l * VOCAB + xs[l]];
    }
    hidH[(size_t)b * HID + h] = (_Float16)tanhf(s);
}

// ------------------------------------------------------------------
// Kernel 2: out = sigmoid(hid @ W2^T + b2) via V_WMMA_F32_16X16X32_F16.
// Grid: one block per 16-wide N tile (7500 blocks), 4 waves/block.
// Wave w accumulates M-tiles [4w, 4w+3] (covers all M=256 per block,
// so W2 streams from HBM exactly once). W2 fp32 -> fp16 in-register.
// ------------------------------------------------------------------
__global__ void __launch_bounds__(128)
nnlm_gemm_sigmoid_kernel(const _Float16* __restrict__ hidH,
                         const float* __restrict__ W2,
                         const float* __restrict__ b2,
                         float* __restrict__ out)
{
    const int lane   = threadIdx.x & 31;
    const int wave   = threadIdx.x >> 5;     // 0..3
    const int n_loc  = lane & 15;            // column within tile / A row within tile
    const int khalf  = lane >> 4;            // 0 or 1: selects contiguous 16-wide K half (this code's order)
    const int n      = blockIdx.x * 16 + n_loc;

    const float* __restrict__ w2row = W2 + (size_t)n * HID;

    const v8f vzero = {0.f, 0.f, 0.f, 0.f, 0.f, 0.f, 0.f, 0.f};
    v8f acc[4];
#pragma unroll
    for (int i = 0; i < 4; ++i) acc[i] = vzero;

    for (int kt = 0; kt < HID / 32; ++kt) {
        const int k0 = kt * 32;

        // ---- B fragment: 16 contiguous K-values of column n (fp32 -> fp16) ----
        // lanes 0-15 hold K = k0..k0+15, lanes 16-31 hold K = k0+16..k0+31
        const float* bsrc = w2row + k0 + khalf * 16;
        v4f f0 = *(const v4f*)(bsrc + 0);
        v4f f1 = *(const v4f*)(bsrc + 4);
        v4f f2 = *(const v4f*)(bsrc + 8);
        v4f f3 = *(const v4f*)(bsrc + 12);
        // speculative prefetch of the next K block of this W2 row
        __builtin_prefetch(bsrc + 32, 0, 0);

        v16h bfrag;
#pragma unroll
        for (int i = 0; i < 4; ++i) {
            bfrag[i]      = (_Float16)f0[i];
            bfrag[4 + i]  = (_Float16)f1[i];
            bfrag[8 + i]  = (_Float16)f2[i];
            bfrag[12 + i] = (_Float16)f3[i];
        }

        // ---- 4 M-tiles per wave ----
#pragma unroll
        for (int mt = 0; mt < 4; ++mt) {
            const int m = (wave * 4 + mt) * 16 + n_loc;   // A row for this lane
            const _Float16* arow = hidH + (size_t)m * HID + k0;
            // contiguous K (matches B): elem i -> K = k0 + 16*khalf + i
            v8h a_lo = *(const v8h*)(arow + khalf * 16);
            v8h a_hi = *(const v8h*)(arow + khalf * 16 + 8);
            v16h afrag = __builtin_shufflevector(a_lo, a_hi,
                0, 1, 2, 3, 4, 5, 6, 7, 8, 9, 10, 11, 12, 13, 14, 15);

            acc[mt] = __builtin_amdgcn_wmma_f32_16x16x32_f16(
                /*neg_a=*/false, afrag,
                /*neg_b=*/false, bfrag,
                /*c_mod=*/(short)0, acc[mt],
                /*reuse_a=*/false, /*reuse_b=*/false);
        }
    }

    // ---- epilogue: + b2, sigmoid (v_exp + v_rcp, no precise-div), store ----
    const float bias = b2[n];
#pragma unroll
    for (int mt = 0; mt < 4; ++mt) {
        const int m_base = (wave * 4 + mt) * 16 + khalf * 8;  // C layout: VGPR j -> row 8*khalf + j
#pragma unroll
        for (int j = 0; j < 8; ++j) {
            const float z = acc[mt][j] + bias;
            const float e = __expf(-z);                       // v_exp_f32
            out[(size_t)(m_base + j) * NOUT + n] = __builtin_amdgcn_rcpf(1.0f + e); // v_rcp_f32
        }
    }
}

extern "C" void kernel_launch(void* const* d_in, const int* in_sizes, int n_in,
                              void* d_out, int out_size, void* d_ws, size_t ws_size,
                              hipStream_t stream) {
    const int*   X  = (const int*)d_in[0];     // (256, 12)
    const float* W1 = (const float*)d_in[1];   // (1024, 120000)
    const float* b1 = (const float*)d_in[2];   // (1024,)
    const float* W2 = (const float*)d_in[3];   // (120000, 1024)
    const float* b2 = (const float*)d_in[4];   // (120000,)
    float* out = (float*)d_out;                // (256, 120000)

    _Float16* hidH = (_Float16*)d_ws;          // 256*1024 fp16 = 512 KB

    nnlm_hidden_kernel<<<BATCH, HID, 0, stream>>>(X, W1, b1, hidH);
    nnlm_gemm_sigmoid_kernel<<<NTILES, 128, 0, stream>>>(hidH, W2, b2, out);
}
